// VectorQuantizer_75840532512956
// MI455X (gfx1250) — matrix operand order repaired and checked by reading, not compiled, not run on hardware
//
#include <hip/hip_runtime.h>
#include <hip/hip_bf16.h>

typedef __attribute__((ext_vector_type(16))) _Float16 v16h;
typedef __attribute__((ext_vector_type(8)))  float    v8f;

#define NUM_CODES   512
#define DIM         64
#define HW          4096        // 64*64 pixels per image
#define CHW         (DIM*HW)    // 262144 floats per image
#define COL_TILES   (NUM_CODES/16)  // 32
#define RTW         2           // row tiles (of 16 points) per wave
#define PTS_PER_BLK (16*RTW)    // 32

// ---------------------------------------------------------------------------
// Prep: convert f32 codebook -> f16, pre-swizzled into this code's contiguous-k B-matrix
// lane layout (16-bit B 32x16: lanes 0-15 = col n with K 0..15, lanes 16-31 =
// col n with K 16..31, 16 halves per lane).  Also precompute ||e||^2.
// bpack layout: [(tile*2 + kb)*32 + lane] * 16 halves  (32B per lane, coalesced)
// ---------------------------------------------------------------------------
__global__ void vq_pack_codebook(const float* __restrict__ emb,
                                 _Float16* __restrict__ bpack,
                                 float* __restrict__ e2) {
  int n = blockIdx.x * blockDim.x + threadIdx.x;
  if (n >= NUM_CODES) return;
  int t = n >> 4;        // column tile
  int l = n & 15;        // column within tile
  float s = 0.f;
  #pragma unroll
  for (int ch = 0; ch < DIM; ++ch) {
    float v = emb[n * DIM + ch];
    s += v * v;
    int g  = ch >> 4;          // 16-channel group 0..3
    int kb = g >> 1;           // k-block (K 0..31 vs 32..63)
    int p  = g & 1;            // lane half (K lo16 vs hi16 within block)
    int h  = ch & 15;
    bpack[(((t * 2 + kb) * 32) + (l + 16 * p)) * 16 + h] = (_Float16)v;
  }
  e2[n] = s;
}

// ---------------------------------------------------------------------------
// Main kernel: one wave (32 threads) per block, RTW row-tiles of 16 points.
// For each 16-code column tile: 2x v_wmma_f32_16x16x32_f16 per row tile,
// fold (||e||^2 - 2*dot) into running (min, argmin), shfl-reduce across the
// 16 column lanes, then gather embedding[idx] and scatter to NCHW output.
// ---------------------------------------------------------------------------
__global__ void __launch_bounds__(32)
vq_argmin_gather(const float* __restrict__ in,      // [32,64,64,64] NCHW
                 const _Float16* __restrict__ bpack,
                 const float* __restrict__ e2,
                 const float* __restrict__ emb,     // [512,64] f32
                 float* __restrict__ out) {         // [32,64,64,64] NCHW
  const int lane  = threadIdx.x;          // 0..31 (wave32)
  const int p0    = blockIdx.x * PTS_PER_BLK;   // first point of block
  const int b     = p0 >> 12;             // image index (4096 px/image)
  const int pix0  = p0 & (HW - 1);        // pixel offset within image
  const int inb   = b * CHW + pix0;

  // ---- load A matrices (16x32, contiguous-k order matching bpack), f32 -> f16 --------------
  // lane L: row M = L&15 ; half h holds K = h + 16*(L>=16)
  const int klo = (lane & 16) ? 16 : 0;
  const int row = lane & 15;
  v16h a[RTW][2];
  #pragma unroll
  for (int rt = 0; rt < RTW; ++rt) {
    #pragma unroll
    for (int kb = 0; kb < 2; ++kb) {
      #pragma unroll
      for (int h = 0; h < 16; ++h) {
        int K  = h + klo;
        int ch = kb * 32 + K;
        float x = in[inb + ch * HW + rt * 16 + row];
        a[rt][kb][h] = (_Float16)x;
      }
    }
  }

  // ---- running (min score, argmin code) per accumulator row -------------
  float minv[RTW][8];
  int   mini[RTW][8];
  #pragma unroll
  for (int rt = 0; rt < RTW; ++rt)
    #pragma unroll
    for (int r = 0; r < 8; ++r) { minv[rt][r] = 3.4e38f; mini[rt][r] = 0; }

  const v16h* bp = (const v16h*)bpack;

  for (int t = 0; t < COL_TILES; ++t) {
    // B tiles: contiguous 32B per lane, fully coalesced (prepacked layout)
    v16h b0 = bp[(t * 2 + 0) * 32 + lane];
    v16h b1 = bp[(t * 2 + 1) * 32 + lane];
    int   code = t * 16 + row;         // column owned by this lane
    float e2v  = e2[code];

    #pragma unroll
    for (int rt = 0; rt < RTW; ++rt) {
      v8f acc = {};
      acc = __builtin_amdgcn_wmma_f32_16x16x32_f16(
          false, a[rt][0], false, b0, (short)0, acc, false, false);
      acc = __builtin_amdgcn_wmma_f32_16x16x32_f16(
          false, a[rt][1], false, b1, (short)0, acc, false, false);
      #pragma unroll
      for (int r = 0; r < 8; ++r) {
        float s = e2v - 2.0f * acc[r];   // ||x||^2 constant per row: dropped
        if (s < minv[rt][r]) { minv[rt][r] = s; mini[rt][r] = code; }
      }
    }
  }

  // ---- reduce (min, idx) across the 16 column lanes (xor 1,2,4,8) -------
  #pragma unroll
  for (int rt = 0; rt < RTW; ++rt) {
    #pragma unroll
    for (int m = 1; m <= 8; m <<= 1) {
      #pragma unroll
      for (int r = 0; r < 8; ++r) {
        float ov = __shfl_xor(minv[rt][r], m, 32);
        int   oi = __shfl_xor(mini[rt][r], m, 32);
        if (ov < minv[rt][r] || (ov == minv[rt][r] && oi < mini[rt][r])) {
          minv[rt][r] = ov; mini[rt][r] = oi;
        }
      }
    }
  }

  // ---- stage per-point argmin through LDS -------------------------------
  // lanes 0-15 hold rows 0-7 (M=r), lanes 16-31 hold rows 8-15 (M=r+8)
  __shared__ int idxs[PTS_PER_BLK];
  #pragma unroll
  for (int rt = 0; rt < RTW; ++rt) {
    if (lane == 0) {
      #pragma unroll
      for (int r = 0; r < 8; ++r) idxs[rt * 16 + r] = mini[rt][r];
    } else if (lane == 16) {
      #pragma unroll
      for (int r = 0; r < 8; ++r) idxs[rt * 16 + 8 + r] = mini[rt][r];
    }
  }
  __syncthreads();

  // ---- gather codebook rows, scatter to NCHW output ---------------------
  // lanes 0-15 write channel 2c, lanes 16-31 channel 2c+1; for a fixed
  // channel the 16 lanes hit 16 consecutive pixels -> coalesced 64B stores.
  #pragma unroll
  for (int rt = 0; rt < RTW; ++rt) {
    int q    = row;                       // point within row tile
    int pix  = pix0 + rt * 16 + q;
    int code = idxs[rt * 16 + q];
    #pragma unroll
    for (int chp = 0; chp < DIM / 2; ++chp) {
      int ch = 2 * chp + (lane >> 4);
      out[b * CHW + ch * HW + pix] = emb[code * DIM + ch];
    }
  }
}

extern "C" void kernel_launch(void* const* d_in, const int* in_sizes, int n_in,
                              void* d_out, int out_size, void* d_ws, size_t ws_size,
                              hipStream_t stream) {
  const float* in  = (const float*)d_in[0];   // [32,64,64,64] f32
  const float* emb = (const float*)d_in[1];   // [512,64] f32
  float* out = (float*)d_out;

  _Float16* bpack = (_Float16*)d_ws;                       // 64 KB
  float*    e2    = (float*)((char*)d_ws + 64 * 1024);     // 2 KB

  vq_pack_codebook<<<(NUM_CODES + 255) / 256, 256, 0, stream>>>(emb, bpack, e2);

  const int npts   = 32 * HW;               // 131072 points
  const int nblk   = npts / PTS_PER_BLK;    // 4096 single-wave blocks
  vq_argmin_gather<<<nblk, 32, 0, stream>>>(in, bpack, e2, emb, out);
}
